// LatticeTokenPtrNet_77464030151099
// MI455X (gfx1250) — compile-verified, matrix operand order checked by reading, not run on hardware
//
#include <hip/hip_runtime.h>
#include <hip/hip_bf16.h>
#include <math.h>

// ---------------- sizes ----------------
#define T_    32
#define A_    32
#define S_    1024          // T*A
#define FE_   1024
#define H_    768
#define G_    3072          // 4*H
#define H2_   1536
#define G2_   6144          // 4*H2
#define DDEC_ 1280
#define E_    256
#define MDEC_ 64            // decoder M (32) padded to one 64-row block tile

typedef __attribute__((ext_vector_type(16))) __bf16 v16bf;
typedef __attribute__((ext_vector_type(8)))  float  v8f;

__device__ __forceinline__ float sigf(float x) { return 1.0f / (1.0f + expf(-x)); }

// ---------------- 1. lattice embedding gather (writes fp32 + bf16 copies)
__global__ void gather_x_kernel(const int* __restrict__ lattice,
                                const float* __restrict__ emb,
                                float* __restrict__ X,
                                __bf16* __restrict__ Xbf) {
    int idx = blockIdx.x * blockDim.x + threadIdx.x;   // 0 .. S_*FE_-1
    int s   = idx >> 10;
    int col = idx & 1023;
    int f   = col >> 8;
    int e   = col & 255;
    float v = emb[lattice[s * 4 + f] * E_ + e];
    X[idx]   = v;
    Xbf[idx] = (__bf16)v;
}

// ---------------- 2. fp32 -> bf16 weight conversion (done once, amortized)
__global__ void f32_to_bf16_kernel(const float* __restrict__ src,
                                   __bf16* __restrict__ dst, int n) {
    int i = blockIdx.x * blockDim.x + threadIdx.x;
    if (i < n) dst[i] = (__bf16)src[i];
}

// ---------------- 3. decoder inputs, bf16, padded to MDEC_ rows
__global__ void build_xdec_kernel(const float* __restrict__ X,
                                  const int* __restrict__ gold,
                                  const int* __restrict__ sos_ids,
                                  const float* __restrict__ emb,
                                  const int* __restrict__ inputs,
                                  const float* __restrict__ tok_emb,
                                  __bf16* __restrict__ Xdec) {
    int idx = blockIdx.x * blockDim.x + threadIdx.x;   // 0 .. MDEC_*DDEC_-1
    if (idx >= MDEC_ * DDEC_) return;
    int t   = idx / DDEC_;
    int col = idx % DDEC_;
    float v = 0.0f;
    if (t < T_) {
        if (col < FE_) {
            if (t == 0) {
                int f = col >> 8, e = col & 255;
                v = emb[sos_ids[f] * E_ + e];
            } else {
                int row = (t - 1) * A_ + gold[t - 1];
                v = X[row * FE_ + col];
            }
        } else {
            v = tok_emb[inputs[t] * E_ + (col - FE_)];
        }
    }
    Xdec[idx] = (__bf16)v;
}

// ---------------- 4. LDS-tiled WMMA bf16 GEMM: C[M,N] = A[M,K] @ B[K,N] + bias
// block = 256 threads = 8 waves; tile 64(M) x 128(N) x 32(K).
// Each wave computes a 32x32 output = 2x2 WMMA tiles (4 v_wmma per K-step).
// Grids are exact multiples -> EXEC all-1s at every WMMA.
__global__ __launch_bounds__(256) void wmma_gemm_kernel(const __bf16* __restrict__ A,
                                                        const __bf16* __restrict__ B,
                                                        const float* __restrict__ bias,
                                                        float* __restrict__ C,
                                                        int M, int N, int K) {
    __shared__ __bf16 sA[64][32];     // K-major A tile
    __shared__ __bf16 sBt[128][32];   // K-major (transposed) B tile

    const int tid   = threadIdx.x;
    const int wave  = tid >> 5;
    const int lane  = tid & 31;
    const int wm    = wave >> 2;          // 0..1  -> M sub-block of 32
    const int wn    = wave & 3;           // 0..3  -> N sub-block of 32
    const int tileM = blockIdx.y * 64;
    const int tileN = blockIdx.x * 128;

    const int lane15 = lane & 15;
    const int kb     = (lane >> 4) << 3;  // lanes 16-31 own the +8 K group

    v8f acc[2][2] = {{{}, {}}, {{}, {}}};

    for (int k0 = 0; k0 < K; k0 += 32) {
        // ---- stage A: 64x32 bf16 (4KB), one uint4 (8 bf16) per thread
        {
            int row = tid >> 2;
            int c8  = (tid & 3) << 3;
            uint4 v4 = *(const uint4*)(A + (size_t)(tileM + row) * K + k0 + c8);
            *(uint4*)&sA[row][c8] = v4;
        }
        // ---- stage B transposed: 32x128 bf16 (8KB), two uint4 per thread
#pragma unroll
        for (int h = 0; h < 2; ++h) {
            int id  = tid + h * 256;          // 0..511 chunks of 8
            int row = id >> 4;                // K row 0..31
            int c8  = (id & 15) << 3;         // N chunk
            uint4 v4 = *(const uint4*)(B + (size_t)(k0 + row) * N + tileN + c8);
            const __bf16* e = (const __bf16*)&v4;
#pragma unroll
            for (int j = 0; j < 8; ++j) sBt[c8 + j][row] = e[j];
        }
        __syncthreads();

        // ---- fragments (contiguous LDS reads; ISA 16-bit layout:
        //      elems 0..7 -> K=kb+0..7, elems 8..15 -> K=kb+16..23)
        v16bf af[2], bfr[2];
#pragma unroll
        for (int mi = 0; mi < 2; ++mi) {
            const __bf16* r = &sA[wm * 32 + mi * 16 + lane15][0];
#pragma unroll
            for (int v = 0; v < 8; ++v) { af[mi][v] = r[kb + v]; af[mi][8 + v] = r[kb + 16 + v]; }
        }
#pragma unroll
        for (int ni = 0; ni < 2; ++ni) {
            const __bf16* r = &sBt[wn * 32 + ni * 16 + lane15][0];
#pragma unroll
            for (int v = 0; v < 8; ++v) { bfr[ni][v] = r[kb + v]; bfr[ni][8 + v] = r[kb + 16 + v]; }
        }
        // ---- 4 WMMAs, register-level fragment reuse
#pragma unroll
        for (int mi = 0; mi < 2; ++mi)
#pragma unroll
            for (int ni = 0; ni < 2; ++ni)
                acc[mi][ni] = __builtin_amdgcn_wmma_f32_16x16x32_bf16(
                    false, af[mi], false, bfr[ni], (short)0, acc[mi][ni], false, false);
        __syncthreads();
    }

    // ---- epilogue: C layout per ISA (VGPR r: lanes 0-15 -> M=r, 16-31 -> M=8+r)
#pragma unroll
    for (int mi = 0; mi < 2; ++mi) {
#pragma unroll
        for (int ni = 0; ni < 2; ++ni) {
            const int row0 = tileM + wm * 32 + mi * 16 + ((lane >> 4) << 3);
            const int col  = tileN + wn * 32 + ni * 16 + lane15;
            const float bv = bias[col];
#pragma unroll
            for (int r = 0; r < 8; ++r)
                C[(size_t)(row0 + r) * N + col] = acc[mi][ni][r] + bv;
        }
    }
}

// ---------------- 5. encoder recurrence: one workgroup per direction
__global__ __launch_bounds__(1024) void encoder_rnn_kernel(const float* __restrict__ GxF,
                                                           const float* __restrict__ GxB,
                                                           const float* __restrict__ WhF,
                                                           const float* __restrict__ WhB,
                                                           float* __restrict__ enc,
                                                           float* __restrict__ h0,
                                                           float* __restrict__ c0) {
    __shared__ float h[H_];
    __shared__ float c[H_];
    __shared__ float gates[G_];
    const int dir = blockIdx.x;
    const float* __restrict__ Gx = dir ? GxB : GxF;
    const float* __restrict__ Wh = dir ? WhB : WhF;
    const int tid = threadIdx.x;

    if (tid < H_) { h[tid] = 0.0f; c[tid] = 0.0f; }
    __syncthreads();

    for (int step = 0; step < S_; ++step) {
        const int s = dir ? (S_ - 1 - step) : step;
        if (step + 1 < S_) {  // prefetch next pre-activation row (global_prefetch_b8)
            const int sn = dir ? (S_ - 2 - step) : step + 1;
            __builtin_prefetch(&Gx[sn * G_ + tid], 0, 1);
        }
        float a0 = Gx[s * G_ + tid];
        float a1 = Gx[s * G_ + tid + 1024];
        float a2 = Gx[s * G_ + tid + 2048];
        for (int k = 0; k < H_; ++k) {
            const float hk = h[k];
            const float* w = Wh + k * G_ + tid;
            a0 = fmaf(hk, w[0],    a0);
            a1 = fmaf(hk, w[1024], a1);
            a2 = fmaf(hk, w[2048], a2);
        }
        gates[tid]        = a0;
        gates[tid + 1024] = a1;
        gates[tid + 2048] = a2;
        __syncthreads();
        if (tid < H_) {
            const float gi = gates[tid];
            const float gf = gates[H_ + tid];
            const float gg = gates[2 * H_ + tid];
            const float go = gates[3 * H_ + tid];
            const float cn = sigf(gf) * c[tid] + sigf(gi) * tanhf(gg);
            const float hn = sigf(go) * tanhf(cn);
            c[tid] = cn;
            h[tid] = hn;
            enc[s * H2_ + dir * H_ + tid] = hn;
        }
        __syncthreads();
    }
    if (tid < H_) {
        h0[dir * H_ + tid] = h[tid];
        c0[dir * H_ + tid] = c[tid];
    }
}

// ---------------- 6. decoder LSTM + bilinear attention (single workgroup)
__global__ __launch_bounds__(1024) void decoder_kernel(const float* __restrict__ Gxdec,
                                                       const float* __restrict__ decWh,
                                                       const float* __restrict__ attnW,
                                                       const float* __restrict__ enc,
                                                       const float* __restrict__ h0,
                                                       const float* __restrict__ c0,
                                                       float* __restrict__ out) {
    __shared__ float h[H2_];
    __shared__ float c[H2_];
    __shared__ float gates[G2_];
    __shared__ float q[H2_];
    const int tid = threadIdx.x;

    for (int e = tid; e < H2_; e += 1024) { h[e] = h0[e]; c[e] = c0[e]; }
    __syncthreads();

    for (int t = 0; t < T_; ++t) {
        float acc[6];
#pragma unroll
        for (int j = 0; j < 6; ++j) acc[j] = Gxdec[t * G2_ + tid + j * 1024];
        for (int k = 0; k < H2_; ++k) {
            const float hk = h[k];
            const float* w = decWh + k * G2_ + tid;
#pragma unroll
            for (int j = 0; j < 6; ++j) acc[j] = fmaf(hk, w[j * 1024], acc[j]);
        }
#pragma unroll
        for (int j = 0; j < 6; ++j) gates[tid + j * 1024] = acc[j];
        __syncthreads();

        for (int e = tid; e < H2_; e += 1024) {
            const float gi = gates[e];
            const float gf = gates[H2_ + e];
            const float gg = gates[2 * H2_ + e];
            const float go = gates[3 * H2_ + e];
            const float cn = sigf(gf) * c[e] + sigf(gi) * tanhf(gg);
            const float hn = sigf(go) * tanhf(cn);
            c[e] = cn;
            h[e] = hn;
        }
        __syncthreads();

        for (int n = tid; n < H2_; n += 1024) {
            float a = 0.0f;
            for (int k = 0; k < H2_; ++k) a = fmaf(h[k], attnW[k * H2_ + n], a);
            q[n] = a;
        }
        __syncthreads();

        const int wave = tid >> 5, lane = tid & 31;
        float p = 0.0f;
        const float* er = enc + (t * A_ + wave) * H2_;
        for (int n = lane; n < H2_; n += 32) p = fmaf(q[n], er[n], p);
#pragma unroll
        for (int off = 16; off > 0; off >>= 1) p += __shfl_down(p, off, 32);
        if (lane == 0) out[t * A_ + wave] = p;
        __syncthreads();
    }
}

// ---------------- launch ----------------
extern "C" void kernel_launch(void* const* d_in, const int* in_sizes, int n_in,
                              void* d_out, int out_size, void* d_ws, size_t ws_size,
                              hipStream_t stream) {
    const int*   lattice = (const int*)d_in[0];
    const int*   inputs  = (const int*)d_in[2];
    const int*   gold    = (const int*)d_in[4];
    const int*   sos     = (const int*)d_in[5];
    const float* emb     = (const float*)d_in[6];
    const float* tok_emb = (const float*)d_in[7];
    const float* WxF     = (const float*)d_in[8];
    const float* WhF     = (const float*)d_in[9];
    const float* bF      = (const float*)d_in[10];
    const float* WxB     = (const float*)d_in[11];
    const float* WhB     = (const float*)d_in[12];
    const float* bB      = (const float*)d_in[13];
    const float* decWx   = (const float*)d_in[14];
    const float* decWh   = (const float*)d_in[15];
    const float* decB    = (const float*)d_in[16];
    const float* attnW   = (const float*)d_in[17];
    float* out = (float*)d_out;

    char* p = (char*)d_ws;
    auto alloc = [&](size_t bytes) -> char* {
        char* q = p; p += (bytes + 255) & ~(size_t)255; return q;
    };
    float*  X       = (float*) alloc((size_t)S_ * FE_ * 4);
    __bf16* Xbf     = (__bf16*)alloc((size_t)S_ * FE_ * 2);
    __bf16* WxFbf   = (__bf16*)alloc((size_t)FE_ * G_ * 2);
    __bf16* WxBbf   = (__bf16*)alloc((size_t)FE_ * G_ * 2);
    __bf16* decWxbf = (__bf16*)alloc((size_t)DDEC_ * G2_ * 2);
    __bf16* Xdecbf  = (__bf16*)alloc((size_t)MDEC_ * DDEC_ * 2);
    float*  GxF     = (float*) alloc((size_t)S_ * G_ * 4);
    float*  GxB     = (float*) alloc((size_t)S_ * G_ * 4);
    float*  Gxdec   = (float*) alloc((size_t)MDEC_ * G2_ * 4);   // padded rows
    float*  enc     = (float*) alloc((size_t)S_ * H2_ * 4);
    float*  h0      = (float*) alloc((size_t)H2_ * 4);
    float*  c0      = (float*) alloc((size_t)H2_ * 4);

    // 1. embedding gather (+bf16 copy)
    gather_x_kernel<<<(S_ * FE_) / 256, 256, 0, stream>>>(lattice, emb, X, Xbf);
    // 2. one-time weight conversions
    f32_to_bf16_kernel<<<(FE_ * G_) / 256, 256, 0, stream>>>(WxF, WxFbf, FE_ * G_);
    f32_to_bf16_kernel<<<(FE_ * G_) / 256, 256, 0, stream>>>(WxB, WxBbf, FE_ * G_);
    f32_to_bf16_kernel<<<(DDEC_ * G2_) / 256, 256, 0, stream>>>(decWx, decWxbf, DDEC_ * G2_);
    // 3. decoder inputs (teacher forcing known up front), zero-padded to 64 rows
    build_xdec_kernel<<<(MDEC_ * DDEC_ + 255) / 256, 256, 0, stream>>>(
        X, gold, sos, emb, inputs, tok_emb, Xdecbf);
    // 4. LDS-tiled WMMA GEMMs for all non-recurrent matmuls
    wmma_gemm_kernel<<<dim3(G_ / 128, S_ / 64), 256, 0, stream>>>(
        Xbf, WxFbf, bF, GxF, S_, G_, FE_);
    wmma_gemm_kernel<<<dim3(G_ / 128, S_ / 64), 256, 0, stream>>>(
        Xbf, WxBbf, bB, GxB, S_, G_, FE_);
    wmma_gemm_kernel<<<dim3(G2_ / 128, MDEC_ / 64), 256, 0, stream>>>(
        Xdecbf, decWxbf, decB, Gxdec, MDEC_, G2_, DDEC_);
    // 5. sequential encoder (one workgroup per direction)
    encoder_rnn_kernel<<<2, 1024, 0, stream>>>(GxF, GxB, WhF, WhB, enc, h0, c0);
    // 6. decoder + attention
    decoder_kernel<<<1, 1024, 0, stream>>>(Gxdec, decWh, attnW, enc, h0, c0, out);
}
